// PhraseModel_75307956568710
// MI455X (gfx1250) — matrix operand order repaired and checked by reading, not compiled, not run on hardware
//
#include <hip/hip_runtime.h>

// VQ nearest-codebook lookup for MI455X (gfx1250, wave32, WMMA).
// argmin_k ||z-c_k||^2 == argmin_k (||c_k||^2 - 2 z.c_k):
// cross term via v_wmma_f32_16x16x32_bf16 (f32 accumulate), epilogue in f32.

typedef __attribute__((ext_vector_type(16))) __bf16 v16bf;
typedef __attribute__((ext_vector_type(8)))  float  v8f;
typedef __attribute__((ext_vector_type(4)))  float  f32x4;
typedef __attribute__((ext_vector_type(2)))  float  f32x2;

#define VQ_K   128
#define VQ_D   510
#define SB     520        // padded bf16 LDS row stride (520*2B=1040B -> bank-conflict-free)
#define TILE_M 128        // z rows per workgroup
#define NT     8          // 8 n-tiles * 16 = 128 codes
#define KSTEPS 16         // 16 k-steps * 32 = 512 (510 zero-padded)

__device__ __forceinline__ unsigned short f2bf(float f) {
  unsigned int u = __float_as_uint(f);
  u += 0x7FFFu + ((u >> 16) & 1u);          // round-to-nearest-even
  return (unsigned short)(u >> 16);
}

union ABPack {
  struct { uint4 lo; uint4 hi; } q;
  v16bf v;
};

__global__ __launch_bounds__(256, 1)
void vq_wmma_kernel(const float* __restrict__ z,
                    const float* __restrict__ cb,
                    float* __restrict__ out) {
  __shared__ __align__(16) unsigned short s_cb[VQ_K * SB];    // bf16 codebook
  __shared__ __align__(16) unsigned short s_z [TILE_M * SB];  // bf16 z tile
  __shared__ float s_csq[VQ_K];
  __shared__ int   s_kidx[TILE_M];

  const int  tid  = threadIdx.x;
  const int  lane = tid & 31;
  const int  wv   = tid >> 5;     // wave 0..7
  const int  lrow = lane & 15;
  const int  half = lane >> 4;
  const long row0 = (long)blockIdx.x * TILE_M;

  // ---- zero padded tail columns d = 510..519 ----
  for (int i = tid; i < VQ_K * (SB - VQ_D); i += 256) {
    int r = i / (SB - VQ_D), c = VQ_D + (i % (SB - VQ_D));
    s_cb[r * SB + c] = 0;
    s_z [r * SB + c] = 0;
  }

  // ---- codebook f32 -> bf16 LDS (coalesced float4 stream, cache-resident) ----
  {
    const f32x4* src = (const f32x4*)cb;                   // 65280 f32, 16B aligned
    for (int i = tid; i < (VQ_K * VQ_D) / 4; i += 256) {
      f32x4 v4 = src[i];
      int e = i * 4;
#pragma unroll
      for (int j = 0; j < 4; ++j) {
        int ee = e + j;
        int r = ee / VQ_D, c = ee - r * VQ_D;
        s_cb[r * SB + c] = f2bf(v4[j]);
      }
    }
  }

  // ---- z tile f32 -> bf16 LDS (coalesced, read-once -> nontemporal) ----
  {
    const f32x4* src = (const f32x4*)(z + row0 * VQ_D);    // row0*510*4 % 16 == 0
    for (int i = tid; i < (TILE_M * VQ_D) / 4; i += 256) {
      f32x4 v4 = __builtin_nontemporal_load(&src[i]);
      int e = i * 4;
#pragma unroll
      for (int j = 0; j < 4; ++j) {
        int ee = e + j;
        int r = ee / VQ_D, c = ee - r * VQ_D;
        s_z[r * SB + c] = f2bf(v4[j]);
      }
    }
  }

  // ---- ||c_k||^2 in f32 from the original codebook ----
  if (tid < VQ_K) {
    const float* crow = cb + (long)tid * VQ_D;
    float s = 0.f;
    for (int d = 0; d < VQ_D; ++d) { float f = crow[d]; s += f * f; }
    s_csq[tid] = s;
  }

  __syncthreads();

  const int m0 = wv * 16;                         // this wave's 16 rows in tile
  const unsigned short* zrow = &s_z[(m0 + lrow) * SB];

  v8f zero = {};
  v8f acc[NT];
#pragma unroll
  for (int nt = 0; nt < NT; ++nt) acc[nt] = zero;

  for (int ks = 0; ks < KSTEPS; ++ks) {
    // A (16x32 bf16): lane(m=lrow,h=half) holds K = 32ks+8h..+7 and 32ks+16+8h..+7
    const int da = ks * 32 + 8 * half;
    ABPack a;
    a.q.lo = *(const uint4*)(zrow + da);
    a.q.hi = *(const uint4*)(zrow + da + 16);
    // B (32x16 bf16): lane(n=lrow,h=half) holds K = 32ks+8h..+7 and 32ks+16+8h..+7
    const int db = ks * 32 + 8 * half;
#pragma unroll
    for (int nt = 0; nt < NT; ++nt) {
      const unsigned short* brow = &s_cb[(nt * 16 + lrow) * SB + db];
      ABPack b;
      b.q.lo = *(const uint4*)(brow);
      b.q.hi = *(const uint4*)(brow + 16);
      acc[nt] = __builtin_amdgcn_wmma_f32_16x16x32_bf16(
          false, a.v, false, b.v, (short)0, acc[nt], false, false);
    }
  }

  // ---- epilogue: load ||c||^2 now (not live across the WMMA loop) ----
  float csq_n[NT];
#pragma unroll
  for (int nt = 0; nt < NT; ++nt) csq_n[nt] = s_csq[nt * 16 + lrow];

  // ---- argmin over 128 codes per row; C layout: VGPR v -> row v+8*half, lane -> col ----
#pragma unroll
  for (int v = 0; v < 8; ++v) {
    float bv = 3.402823466e38f;
    int   bi = 0;
#pragma unroll
    for (int nt = 0; nt < NT; ++nt) {
      float s  = csq_n[nt] - 2.0f * acc[nt][v];
      int  idx = nt * 16 + lrow;
      if (s < bv || (s == bv && idx < bi)) { bv = s; bi = idx; }
    }
#pragma unroll
    for (int m = 1; m < 16; m <<= 1) {       // stays within 16-lane half
      float ov = __shfl_xor(bv, m, 32);
      int   oi = __shfl_xor(bi, m, 32);
      if (ov < bv || (ov == bv && oi < bi)) { bv = ov; bi = oi; }
    }
    if (lrow == 0) s_kidx[m0 + v + 8 * half] = bi;
  }
  __syncthreads();

  // ---- gather: out[row] = codebook[kidx[row]] in f32 (codebook L2-hot, out write-once) ----
  for (int r = 0; r < 16; ++r) {
    int k = s_kidx[m0 + r];
    const f32x2* src = (const f32x2*)(cb + (long)k * VQ_D);          // 8B aligned
    f32x2*       dst = (f32x2*)(out + (row0 + m0 + r) * (long)VQ_D); // 8B aligned
#pragma unroll
    for (int i = 0; i < 8; ++i) {
      int c2 = lane + 32 * i;
      if (c2 < VQ_D / 2)
        __builtin_nontemporal_store(src[c2], &dst[c2]);  // 255 float2 = 510 floats
    }
  }
}

extern "C" void kernel_launch(void* const* d_in, const int* in_sizes, int n_in,
                              void* d_out, int out_size, void* d_ws, size_t ws_size,
                              hipStream_t stream) {
  const float* z  = (const float*)d_in[0];   // [B, 510] f32
  const float* cb = (const float*)d_in[1];   // [128, 510] f32
  float* out = (float*)d_out;                // [B, 510] f32
  const int Brows = in_sizes[0] / VQ_D;      // 262144
  dim3 grid(Brows / TILE_M), block(256);
  vq_wmma_kernel<<<grid, block, 0, stream>>>(z, cb, out);
  (void)n_in; (void)out_size; (void)d_ws; (void)ws_size;
}
